// GeniePath_67705864454155
// MI455X (gfx1250) — matrix operand order repaired and checked by reading, not compiled, not run on hardware
//
#include <hip/hip_runtime.h>
#include <hip/hip_bf16.h>
#include <math.h>

// ---------------------------------------------------------------------------
// GeniePath layer for MI455X (gfx1250, wave32, WMMA bf16 16x16x32, f32 accum)
//   N=100000 nodes, E=1.6M edges, D=128, K=3 heads, H=64, HID=192
// ---------------------------------------------------------------------------

typedef __attribute__((ext_vector_type(16))) __bf16 v16bf;
typedef __attribute__((ext_vector_type(8)))  __bf16 v8bf;
typedef __attribute__((ext_vector_type(8)))  float  v8f;

// ---- WMMA fragment loaders (per CDNA5 ISA 16-bit VGPR layouts) -------------
// A 16x32: lane<16 -> row=lane, k in [kb,kb+8) and [kb+16,kb+24), kb=kk*32+hi*8
__device__ __forceinline__ v16bf ld_a_frag(const __bf16* row, int kb) {
  union { v16bf v; v8bf h[2]; } u;
  u.h[0] = *(const v8bf*)(row + kb);
  u.h[1] = *(const v8bf*)(row + kb + 16);
  return u.v;
}
// B 32x16: lane holds col=lane&15, k in [kb,kb+8) and [kb+16,kb+24), kb=kk*32+hi*8
__device__ __forceinline__ v16bf ld_b_frag(const __bf16* col, int kb) {
  union { v16bf v; v8bf h[2]; } u;
  u.h[0] = *(const v8bf*)(col + kb);
  u.h[1] = *(const v8bf*)(col + kb + 16);
  return u.v;
}

__device__ __forceinline__ void atomicMaxF(float* a, float v) {
  if (v >= 0.0f) atomicMax((int*)a, __float_as_int(v));
  else           atomicMin((unsigned int*)a, __float_as_uint(v));
}

__device__ __forceinline__ float sigmoidf_(float v) {
  return 1.0f / (1.0f + __expf(-v));
}

// ---------------------------------------------------------------------------
// Weight prep: Bt1[n=kh*64+h][d] = W[kh][d][h]  (bf16, K-major per column)
//              Wt2[cc=3j+t][k]   = W_ih[{j, 384+j, 576+j}[t]][k]  (f-gate dead)
//              b2[cc] = b_ih[r] + b_hh[r]
// ---------------------------------------------------------------------------
__global__ void k_prep(const float* __restrict__ W, const float* __restrict__ Wih,
                       const float* __restrict__ bih, const float* __restrict__ bhh,
                       __bf16* __restrict__ bt1, __bf16* __restrict__ wt2,
                       float* __restrict__ b2) {
  int i = blockIdx.x * blockDim.x + threadIdx.x;
  if (i < 192 * 128) {
    int n = i >> 7, d = i & 127;
    int kh = n >> 6, h = n & 63;
    bt1[i] = (__bf16)W[(kh * 128 + d) * 64 + h];
  }
  if (i < 576 * 192) {
    int cc = i / 192, k = i - cc * 192;
    int j = cc / 3, t = cc - 3 * j;
    int r = (t == 0) ? j : ((t == 1) ? 384 + j : 576 + j);
    wt2[i] = (__bf16)Wih[r * 192 + k];
  }
  if (i < 576) {
    int j = i / 3, t = i - 3 * j;
    int r = (t == 0) ? j : ((t == 1) ? 384 + j : 576 + j);
    b2[i] = bih[r] + bhh[r];
  }
}

// ---------------------------------------------------------------------------
// GEMM1: hp[N][192] = x[N][128] @ Bt1^T   (bf16 WMMA, f32 accumulate)
// block: 256 thr = 8 waves * 16 rows = 128 rows; 96 cols per block (grid.y=2)
// ---------------------------------------------------------------------------
__global__ __launch_bounds__(256) void k_gemm1(const float* __restrict__ x,
                                               const __bf16* __restrict__ bt1,
                                               float* __restrict__ hp, int nN) {
  __shared__ __bf16 xs[128][136];  // padded: 272B rows -> conflict-free, 16B aligned
  __shared__ __bf16 wt[96][136];
  const int t = threadIdx.x;
  const int row0 = blockIdx.x * 128;
  const int col0 = blockIdx.y * 96;

  for (int i = t; i < 96 * 128; i += 256) {
    int c = i >> 7, d = i & 127;
    wt[c][d] = bt1[(col0 + c) * 128 + d];
  }
  for (int i = t; i < 128 * 128; i += 256) {
    int r = i >> 7, d = i & 127;
    int gr = row0 + r; if (gr >= nN) gr = nN - 1;
    xs[r][d] = (__bf16)x[gr * 128 + d];
  }
  __syncthreads();

  const int wave = t >> 5, lane = t & 31;
  const int lm = lane & 15, hi = lane >> 4;
  const __bf16* arow = &xs[wave * 16 + lm][0];

  v8f acc[6];
#pragma unroll
  for (int ct = 0; ct < 6; ++ct)
#pragma unroll
    for (int j = 0; j < 8; ++j) acc[ct][j] = 0.0f;

#pragma unroll
  for (int kk = 0; kk < 4; ++kk) {           // K = 128 = 4 * 32
    v16bf a = ld_a_frag(arow, kk * 32 + hi * 8);
#pragma unroll
    for (int ct = 0; ct < 6; ++ct) {
      v16bf b = ld_b_frag(&wt[ct * 16 + lm][0], kk * 32 + hi * 8);
      acc[ct] = __builtin_amdgcn_wmma_f32_16x16x32_bf16(
          false, a, false, b, (short)0, acc[ct], false, false);
    }
  }

#pragma unroll
  for (int ct = 0; ct < 6; ++ct)
#pragma unroll
    for (int j = 0; j < 8; ++j) {            // C/D: vgpr j -> m = j + hi*8
      int gr = row0 + wave * 16 + hi * 8 + j;
      if (gr < nN) hp[gr * 192 + col0 + ct * 16 + lm] = acc[ct][j];
    }
}

// ---------------------------------------------------------------------------
// Per-node attention dot products: sd[n,k] = hp[n,k,:].att[k,:64]
//                                  ss[n,k] = hp[n,k,:].att[k,64:]
// ---------------------------------------------------------------------------
__global__ void k_scores(const float* __restrict__ hp, const float* __restrict__ att,
                         float* __restrict__ sd, float* __restrict__ ss, int nN) {
  int i = blockIdx.x * blockDim.x + threadIdx.x;
  if (i >= nN * 3) return;
  int n = i / 3, k = i - 3 * n;
  const float* row = hp + n * 192 + k * 64;
  const float* a1 = att + k * 128;
  const float* a2 = a1 + 64;
  float d0 = 0.0f, s0 = 0.0f;
#pragma unroll 8
  for (int h = 0; h < 64; ++h) { float v = row[h]; d0 += v * a1[h]; s0 += v * a2[h]; }
  sd[i] = d0; ss[i] = s0;
}

__global__ void k_init(float* __restrict__ mx, float* __restrict__ sum, int n3) {
  int i = blockIdx.x * blockDim.x + threadIdx.x;
  if (i < n3) { mx[i] = -__builtin_inff(); sum[i] = 0.0f; }
}

// Edge pass 1: leaky-relu score, stash, segment-max into mx[dst,k]
__global__ void k_edge1(const int* __restrict__ src, const int* __restrict__ dst,
                        const float* __restrict__ sd, const float* __restrict__ ss,
                        float* __restrict__ se, float* __restrict__ mx, int nE) {
  int e = blockIdx.x * blockDim.x + threadIdx.x;
  if (e >= nE) return;
  int s = src[e], d = dst[e];
#pragma unroll
  for (int k = 0; k < 3; ++k) {
    float v = sd[d * 3 + k] + ss[s * 3 + k];
    v = (v > 0.0f) ? v : 0.01f * v;           // leaky_relu(0.01)
    se[e * 3 + k] = v;
    atomicMaxF(&mx[d * 3 + k], v);
  }
}

// Edge pass 2: segment-sum of exp(score - max)
__global__ void k_edge2(const int* __restrict__ dst, const float* __restrict__ se,
                        const float* __restrict__ mx, float* __restrict__ sum, int nE) {
  int e = blockIdx.x * blockDim.x + threadIdx.x;
  if (e >= nE) return;
  int d = dst[e];
#pragma unroll
  for (int k = 0; k < 3; ++k)
    atomicAdd(&sum[d * 3 + k], __expf(se[e * 3 + k] - mx[d * 3 + k]));
}

// h_agg = tanh(hp * (s / max(s,1e-9))), packed bf16 pairs (GEMM2 A-matrix)
__global__ void k_hagg(const float* __restrict__ hp, const float* __restrict__ sum,
                       unsigned int* __restrict__ hagg_u32, int nN) {
  int i = blockIdx.x * blockDim.x + threadIdx.x;     // over N*96 bf16 pairs
  if (i >= nN * 96) return;
  int n = i / 96, p = i - n * 96;
  int c = 2 * p;                                      // pairs never cross head (64 even)
  float s = sum[n * 3 + (c >> 6)];
  float f = s / fmaxf(s, 1e-9f);                      // 1 if node has in-edges else 0
  union { unsigned int u; __bf16 b[2]; } o;
  o.b[0] = (__bf16)tanhf(hp[n * 192 + c]     * f);
  o.b[1] = (__bf16)tanhf(hp[n * 192 + c + 1] * f);
  hagg_u32[n * 96 + p] = o.u;
}

// ---------------------------------------------------------------------------
// GEMM2 + LSTM epilogue: gates(i,g,o) = hagg @ Wt2^T + b2, then
//   h_new = sigmoid(o) * tanh(sigmoid(i) * tanh(g))      (f-gate, W_hh dead)
// block: 128 thr = 4 waves * 16 rows = 64 rows; 96 gate-cols = 32 outputs
// ---------------------------------------------------------------------------
__global__ __launch_bounds__(128) void k_gemm2(const __bf16* __restrict__ hagg,
                                               const __bf16* __restrict__ wt2,
                                               const float* __restrict__ b2,
                                               float* __restrict__ out, int nN) {
  __shared__ __align__(16) char smem[64000];
  __bf16 (*xs)[200] = (__bf16 (*)[200])smem;            // 64*400B = 25600
  __bf16 (*wt)[200] = (__bf16 (*)[200])(smem + 25600);  // 96*400B = 38400
  float  (*gl)[96]  = (float  (*)[96])smem;             // 24576B, aliases xs (post-GEMM)

  const int t = threadIdx.x;
  const int row0 = blockIdx.x * 64;
  const int cc0 = blockIdx.y * 96;

  for (int i = t; i < 64 * 96; i += 128) {              // A tile: bf16 pairs
    int r = i / 96, p = i - r * 96;
    int gr = row0 + r; if (gr >= nN) gr = nN - 1;
    *(unsigned int*)&xs[r][2 * p] = *(const unsigned int*)(hagg + gr * 192 + 2 * p);
  }
  for (int i = t; i < 96 * 96; i += 128) {              // B tile
    int c = i / 96, p = i - c * 96;
    *(unsigned int*)&wt[c][2 * p] = *(const unsigned int*)(wt2 + (cc0 + c) * 192 + 2 * p);
  }
  __syncthreads();

  const int wave = t >> 5, lane = t & 31;
  const int lm = lane & 15, hi = lane >> 4;
  const __bf16* arow = &xs[wave * 16 + lm][0];

  v8f acc[6];
#pragma unroll
  for (int ct = 0; ct < 6; ++ct)
#pragma unroll
    for (int j = 0; j < 8; ++j) acc[ct][j] = 0.0f;

#pragma unroll
  for (int kk = 0; kk < 6; ++kk) {                      // K = 192 = 6 * 32
    v16bf a = ld_a_frag(arow, kk * 32 + hi * 8);
#pragma unroll
    for (int ct = 0; ct < 6; ++ct) {
      v16bf b = ld_b_frag(&wt[ct * 16 + lm][0], kk * 32 + hi * 8);
      acc[ct] = __builtin_amdgcn_wmma_f32_16x16x32_bf16(
          false, a, false, b, (short)0, acc[ct], false, false);
    }
  }
  __syncthreads();                                      // xs reads done -> reuse as gl

#pragma unroll
  for (int ct = 0; ct < 6; ++ct) {
    float bias = b2[cc0 + ct * 16 + lm];
#pragma unroll
    for (int j = 0; j < 8; ++j)
      gl[wave * 16 + hi * 8 + j][ct * 16 + lm] = acc[ct][j] + bias;
  }
  __syncthreads();

  const int j0 = blockIdx.y * 32;                       // 32 hidden cols per block
  for (int i = t; i < 64 * 32; i += 128) {
    int m = i >> 5, jl = i & 31;
    int gr = row0 + m;
    if (gr < nN) {
      float gi = gl[m][3 * jl + 0];
      float gg = gl[m][3 * jl + 1];
      float go = gl[m][3 * jl + 2];
      float c = sigmoidf_(gi) * tanhf(gg);
      out[gr * 192 + j0 + jl] = sigmoidf_(go) * tanhf(c);
    }
  }
}

// ---------------------------------------------------------------------------
static size_t align256(size_t v) { return (v + 255) & ~(size_t)255; }

extern "C" void kernel_launch(void* const* d_in, const int* in_sizes, int n_in,
                              void* d_out, int out_size, void* d_ws, size_t ws_size,
                              hipStream_t stream) {
  const float* x   = (const float*)d_in[0];
  const int*   src = (const int*)d_in[1];
  const int*   dst = (const int*)d_in[2];
  const float* W   = (const float*)d_in[3];
  const float* att = (const float*)d_in[4];
  const float* Wih = (const float*)d_in[5];
  const float* bih = (const float*)d_in[6];
  /* d_in[7] = W_hh: dead (h0 == 0) */
  const float* bhh = (const float*)d_in[8];
  float* out = (float*)d_out;

  const int nN = in_sizes[0] / 128;   // 100000
  const int nE = in_sizes[1];         // 1600000

  // workspace layout
  char* ws = (char*)d_ws;
  size_t off = 0;
  float*  hp   = (float*)(ws + off); off = align256(off + (size_t)nN * 192 * 4);
  __bf16* hagg = (__bf16*)(ws + off); off = align256(off + (size_t)nN * 192 * 2);
  float*  sd   = (float*)(ws + off); off = align256(off + (size_t)nN * 3 * 4);
  float*  ss   = (float*)(ws + off); off = align256(off + (size_t)nN * 3 * 4);
  float*  mx   = (float*)(ws + off); off = align256(off + (size_t)nN * 3 * 4);
  float*  sum  = (float*)(ws + off); off = align256(off + (size_t)nN * 3 * 4);
  float*  se   = (float*)(ws + off); off = align256(off + (size_t)nE * 3 * 4);
  __bf16* bt1  = (__bf16*)(ws + off); off = align256(off + (size_t)192 * 128 * 2);
  __bf16* wt2  = (__bf16*)(ws + off); off = align256(off + (size_t)576 * 192 * 2);
  float*  b2   = (float*)(ws + off); off = align256(off + (size_t)576 * 4);

  k_prep<<<(576 * 192 + 255) / 256, 256, 0, stream>>>(W, Wih, bih, bhh, bt1, wt2, b2);

  k_gemm1<<<dim3((nN + 127) / 128, 2), 256, 0, stream>>>(x, bt1, hp, nN);

  k_scores<<<(nN * 3 + 255) / 256, 256, 0, stream>>>(hp, att, sd, ss, nN);

  k_init<<<(nN * 3 + 255) / 256, 256, 0, stream>>>(mx, sum, nN * 3);

  k_edge1<<<(nE + 255) / 256, 256, 0, stream>>>(src, dst, sd, ss, se, mx, nE);

  k_edge2<<<(nE + 255) / 256, 256, 0, stream>>>(dst, se, mx, sum, nE);

  k_hagg<<<(nN * 96 + 255) / 256, 256, 0, stream>>>(hp, sum, (unsigned int*)hagg, nN);

  k_gemm2<<<dim3((nN + 63) / 64, 6), 128, 0, stream>>>(hagg, wt2, b2, out, nN);
}
